// DEQFixedPoint_42984032698388
// MI455X (gfx1250) — compile-verified, matrix operand order checked by reading, not run on hardware
//
#include <hip/hip_runtime.h>

#define DIM 4096
#define BSZ 64
#define MEM 5
#define LAM 1e-4f
#define TOL 0.01f

typedef __attribute__((ext_vector_type(16))) __bf16 v16bf;
typedef __attribute__((ext_vector_type(8)))  float  v8f;
typedef __attribute__((ext_vector_type(4)))  unsigned int u32x4;
typedef __attribute__((ext_vector_type(8)))  unsigned int u32x8;

union FragBF { v16bf v; uint4 q[2]; };

static __device__ __forceinline__ unsigned short f2bf(float f) {
    union { float f; unsigned u; } v; v.f = f;
    unsigned r = v.u + 0x7FFFu + ((v.u >> 16) & 1u);   // round-to-nearest-even
    return (unsigned short)(r >> 16);
}

// ---------------------------------------------------------------- W -> bf16, transposed
// WbT[n][k] = bf16(W[k][n]); 64x64 tiles through LDS, coalesced both sides.
__global__ __launch_bounds__(256) void convw_t_kernel(const float* __restrict__ W,
                                                      unsigned short* __restrict__ WbT) {
    __shared__ float tile[64][65];                   // +1 pad: conflict-free column reads
    int tid = threadIdx.x;
    int n0 = blockIdx.x * 64, k0 = blockIdx.y * 64;
#pragma unroll
    for (int it = 0; it < 4; ++it) {
        int e = (it * 256 + tid) * 4;                // 4096 floats
        int r = e >> 6, c = e & 63;                  // r: k, c: n
        float4 f = *(const float4*)&W[(size_t)(k0 + r) * DIM + n0 + c];
        tile[r][c] = f.x; tile[r][c + 1] = f.y; tile[r][c + 2] = f.z; tile[r][c + 3] = f.w;
    }
    __syncthreads();
#pragma unroll
    for (int it = 0; it < 2; ++it) {
        int e = (it * 256 + tid) * 8;                // 4096 bf16
        int n = e >> 6, kk = e & 63;
        unsigned short o[8];
#pragma unroll
        for (int i = 0; i < 8; ++i) o[i] = f2bf(tile[kk + i][n]);
        *(uint4*)&WbT[(size_t)(n0 + n) * DIM + k0 + kk] = *(uint4*)o;
    }
}

// ---------------------------------------------------------------- init: F0 = tanh(x+b)
__global__ __launch_bounds__(256) void init_kernel(const float* __restrict__ x,
                                                   const float* __restrict__ bias,
                                                   float* __restrict__ Xh,
                                                   float* __restrict__ Fh,
                                                   unsigned short* __restrict__ Zb,
                                                   int* flag, int* last_idx) {
    int i = blockIdx.x * 256 + threadIdx.x;          // over BSZ*DIM
    if (i == 0) { *flag = 0; *last_idx = 1; }
    int b = i >> 12, d = i & (DIM - 1);
    float f0 = tanhf(x[i] + bias[d]);
    size_t base = ((size_t)b * MEM) * DIM + d;
    Xh[base + 0 * DIM] = 0.f;
    Xh[base + 1 * DIM] = f0;
    Xh[base + 2 * DIM] = 0.f;
    Xh[base + 3 * DIM] = 0.f;
    Xh[base + 4 * DIM] = 0.f;
    Fh[base + 0 * DIM] = f0;
    Fh[base + 1 * DIM] = 0.f;
    Fh[base + 2 * DIM] = 0.f;
    Fh[base + 3 * DIM] = 0.f;
    Fh[base + 4 * DIM] = 0.f;
    Zb[i] = f2bf(f0);
}

// ---------------------------------------------------------------- WMMA GEMM + tanh epilogue
// F[:, fslot] = tanh(Zb @ W + x + b), W given pre-transposed bf16 (WbT[n][k]).
// 64 blocks x 256 threads (8 waves). Block: 64 rows x 64 cols.
// A tile (Z 64x32) arrives via Tensor Data Mover; B tile via coalesced b128.
__global__ __launch_bounds__(256) void gemm_tanh_kernel(const unsigned short* __restrict__ Zb,
                                                        const unsigned short* __restrict__ WbT,
                                                        const float* __restrict__ x,
                                                        const float* __restrict__ bias,
                                                        float* __restrict__ Fh,
                                                        int fslot,
                                                        const int* __restrict__ flag) {
    if (*flag) return;
    __shared__ unsigned short As[BSZ * 32];   // [row][k]  4 KB  (TDM destination)
    __shared__ unsigned short Bs[64 * 32];    // [n][k]    4 KB

    const int tid   = threadIdx.x;
    const int lane  = tid & 31;
    const int wave  = tid >> 5;
    const int half  = lane >> 4;          // 0/1
    const int l16   = lane & 15;
    const int c0    = blockIdx.x * 64;    // N chunk base
    const int mtile = wave >> 1;          // 0..3
    const int ngrp  = wave & 1;           // ntiles 2*ngrp, 2*ngrp+1

    const unsigned lds_a = (unsigned)(size_t)&As[0];   // flat low 32 bits == LDS byte offset

    v8f acc[2] = {};

    for (int k0 = 0; k0 < DIM; k0 += 32) {
        __syncthreads();                  // previous fragments consumed
        if (wave == 0) {
            // TDM D#: 2-D tensor, data_size=2B, tile 32(k) x 64(rows), stride 4096
            unsigned long long ga = (unsigned long long)(size_t)(Zb + k0);
            u32x4 g0 = { 1u,                                   // count=1
                         lds_a,                                // lds_addr
                         (unsigned)ga,                         // global_addr[31:0]
                         ((unsigned)(ga >> 32) & 0x01FFFFFFu) | 0x80000000u }; // ga[56:32] | type=2
            u32x8 g1 = { 0x00010000u,     // data_size=1 (2 bytes), mask=0
                         0x10000000u,     // tensor_dim0=4096 -> bits[79:48] low half
                         0x00400000u,     // tensor_dim1=64   -> bits[111:80] low half
                         0x00200000u,     // tile_dim0=32     -> bits[127:112]
                         0x00000040u,     // tile_dim1=64     -> bits[143:128]
                         4096u,           // tensor_dim0_stride[31:0]
                         0u, 0u };
            asm volatile("tensor_load_to_lds %0, %1" :: "s"(g0), "s"(g1) : "memory");
            __builtin_amdgcn_s_wait_tensorcnt(0);
        }
        // B tile 64(n) x 32(k), K-contiguous: one b128 per thread
        {
            int e = tid * 8;                       // 2048 bf16 total
            int n = e >> 5, kk = e & 31;
            *(uint4*)&Bs[e] = *(const uint4*)&WbT[(size_t)(c0 + n) * DIM + k0 + kk];
            if (k0 + 32 < DIM)
                __builtin_prefetch(&WbT[(size_t)(c0 + n) * DIM + k0 + 32 + kk], 0, 1);
        }
        __syncthreads();

        // A fragment (16x32 bf16): VGPR0..3 = K 8*half.., VGPR4..7 = +16
        FragBF a;
        {
            int row = mtile * 16 + l16;
            a.q[0] = *(const uint4*)&As[row * 32 + 8 * half];
            a.q[1] = *(const uint4*)&As[row * 32 + 16 + 8 * half];
        }
#pragma unroll
        for (int t = 0; t < 2; ++t) {
            int nl = (ngrp * 2 + t) * 16 + l16;
            FragBF bf;
            bf.q[0] = *(const uint4*)&Bs[nl * 32 + 8 * half];
            bf.q[1] = *(const uint4*)&Bs[nl * 32 + 16 + 8 * half];
            acc[t] = __builtin_amdgcn_wmma_f32_16x16x32_bf16(
                false, a.v, false, bf.v, (short)0, acc[t], false, false);
        }
    }

    // C/D layout: VGPR r -> M = mtile*16 + 8*half + r ; N = l16
#pragma unroll
    for (int t = 0; t < 2; ++t) {
        int col = c0 + (ngrp * 2 + t) * 16 + l16;
        float bv = bias[col];
#pragma unroll
        for (int r = 0; r < 8; ++r) {
            int row = mtile * 16 + 8 * half + r;
            float v = acc[t][r] + x[(size_t)row * DIM + col] + bv;
            Fh[((size_t)row * MEM + fslot) * DIM + col] = tanhf(v);
        }
    }
}

// ---------------------------------------------------------------- Gram: GG[b] = G G^T (masked)
__global__ __launch_bounds__(256) void gram_kernel(const float* __restrict__ Xh,
                                                   const float* __restrict__ Fh,
                                                   float* __restrict__ GG,
                                                   int n, const int* __restrict__ flag) {
    if (*flag) return;
    const int PI[15] = {0,0,0,0,0,1,1,1,1,2,2,2,3,3,4};
    const int PJ[15] = {0,1,2,3,4,1,2,3,4,2,3,4,3,4,4};
    int b = blockIdx.x, tid = threadIdx.x;
    int lane = tid & 31, wave = tid >> 5;
    const float* Xb = Xh + (size_t)b * MEM * DIM;
    const float* Fb = Fh + (size_t)b * MEM * DIM;
    float s[15];
#pragma unroll
    for (int p = 0; p < 15; ++p) s[p] = 0.f;
    for (int d = tid; d < DIM; d += 256) {
        float g[MEM];
#pragma unroll
        for (int j = 0; j < MEM; ++j)
            g[j] = (j < n) ? (Fb[j * DIM + d] - Xb[j * DIM + d]) : 0.f;
#pragma unroll
        for (int p = 0; p < 15; ++p) s[p] += g[PI[p]] * g[PJ[p]];
    }
    __shared__ float red[15 * 8];
#pragma unroll
    for (int p = 0; p < 15; ++p) {
        float v = s[p];
        for (int off = 16; off > 0; off >>= 1) v += __shfl_down(v, off, 32);
        if (lane == 0) red[p * 8 + wave] = v;
    }
    __syncthreads();
    if (tid < 15) {
        float t = 0.f;
        for (int w = 0; w < 8; ++w) t += red[tid * 8 + w];
        int i = PI[tid], j = PJ[tid];
        GG[b * 25 + i * 5 + j] = t;
        GG[b * 25 + j * 5 + i] = t;
    }
}

// ---------------------------------------------------------------- solve 6x6 per batch row
__global__ void solve_kernel(const float* __restrict__ GG, float* __restrict__ alpha,
                             int n, const int* __restrict__ flag) {
    if (*flag) return;
    int b = blockIdx.x * blockDim.x + threadIdx.x;
    if (b >= BSZ) return;
    float H[6][7];
    for (int i = 0; i < 6; ++i)
        for (int j = 0; j < 7; ++j) H[i][j] = 0.f;
    H[0][6] = 1.f;                                   // rhs = e0
    for (int j = 0; j < MEM; ++j) {
        float mj = (j < n) ? 1.f : 0.f;
        H[0][j + 1] = mj; H[j + 1][0] = mj;
    }
    for (int i = 0; i < MEM; ++i)
        for (int j = 0; j < MEM; ++j) {
            float v;
            if (i < n && j < n) v = GG[b * 25 + i * 5 + j] + ((i == j) ? LAM : 0.f);
            else                v = (i == j) ? 1.f : 0.f;
            H[i + 1][j + 1] = v;
        }
    // Gauss-Jordan with partial pivoting (H[0][0]==0 -> pivoting required)
    for (int c = 0; c < 6; ++c) {
        int piv = c; float best = fabsf(H[c][c]);
        for (int r = c + 1; r < 6; ++r) {
            float a = fabsf(H[r][c]);
            if (a > best) { best = a; piv = r; }
        }
        if (piv != c)
            for (int j = 0; j < 7; ++j) { float t = H[c][j]; H[c][j] = H[piv][j]; H[piv][j] = t; }
        float inv = 1.f / H[c][c];
        for (int j = 0; j < 7; ++j) H[c][j] *= inv;
        for (int r = 0; r < 6; ++r) {
            if (r == c) continue;
            float f = H[r][c];
            for (int j = 0; j < 7; ++j) H[r][j] -= f * H[c][j];
        }
    }
    for (int j = 0; j < MEM; ++j) alpha[b * MEM + j] = H[j + 1][6];
}

// ---------------------------------------------------------------- Xk = sum_j alpha_j F_j (beta=1)
__global__ __launch_bounds__(256) void update_kernel(const float* __restrict__ Fh,
                                                     const float* __restrict__ alpha,
                                                     float* __restrict__ Xh,
                                                     unsigned short* __restrict__ Zb,
                                                     int idx, const int* __restrict__ flag) {
    if (*flag) return;
    int i = blockIdx.x * 256 + threadIdx.x;          // over BSZ*DIM
    int b = i >> 12, d = i & (DIM - 1);
    const float* ab = alpha + b * MEM;
    const float* Fb = Fh + (size_t)b * MEM * DIM;
    float xk = 0.f;
#pragma unroll
    for (int j = 0; j < MEM; ++j) xk += ab[j] * Fb[j * DIM + d];
    Xh[((size_t)b * MEM + idx) * DIM + d] = xk;
    Zb[i] = f2bf(xk);
}

// ---------------------------------------------------------------- residual partials
__global__ __launch_bounds__(256) void resid_partial_kernel(const float* __restrict__ Xh,
                                                            const float* __restrict__ Fh,
                                                            float* __restrict__ pnum,
                                                            float* __restrict__ pden,
                                                            int idx, const int* __restrict__ flag) {
    if (*flag) return;
    int tid = threadIdx.x, lane = tid & 31, wave = tid >> 5;
    float num = 0.f, den = 0.f;
    for (int i = blockIdx.x * 256 + tid; i < BSZ * DIM; i += 512 * 256) {
        int b = i >> 12, d = i & (DIM - 1);
        size_t o = ((size_t)b * MEM + idx) * DIM + d;
        float fk = Fh[o], xk = Xh[o];
        float df = fk - xk;
        num += df * df; den += fk * fk;
    }
    for (int off = 16; off > 0; off >>= 1) {
        num += __shfl_down(num, off, 32);
        den += __shfl_down(den, off, 32);
    }
    __shared__ float sn[8], sd[8];
    if (lane == 0) { sn[wave] = num; sd[wave] = den; }
    __syncthreads();
    if (tid == 0) {
        float a = 0.f, c = 0.f;
        for (int w = 0; w < 8; ++w) { a += sn[w]; c += sd[w]; }
        pnum[blockIdx.x] = a; pden[blockIdx.x] = c;
    }
}

__global__ __launch_bounds__(512) void resid_final_kernel(const float* __restrict__ pnum,
                                                          const float* __restrict__ pden,
                                                          int* flag, int* last_idx, int idx) {
    if (*flag) return;
    int tid = threadIdx.x, lane = tid & 31, wave = tid >> 5;
    float num = pnum[tid], den = pden[tid];
    for (int off = 16; off > 0; off >>= 1) {
        num += __shfl_down(num, off, 32);
        den += __shfl_down(den, off, 32);
    }
    __shared__ float sn[16], sd[16];
    if (lane == 0) { sn[wave] = num; sd[wave] = den; }
    __syncthreads();
    if (tid == 0) {
        float a = 0.f, c = 0.f;
        for (int w = 0; w < 16; ++w) { a += sn[w]; c += sd[w]; }
        float res = sqrtf(a) / (1e-5f + sqrtf(c));
        *last_idx = idx;
        if (res < TOL) *flag = 1;
    }
}

// ---------------------------------------------------------------- out = F[:, last_idx] == f(z*)
__global__ __launch_bounds__(256) void copy_out_kernel(const float* __restrict__ Fh,
                                                       const int* __restrict__ last_idx,
                                                       float* __restrict__ out) {
    int idx = *last_idx;
    int i = blockIdx.x * 256 + threadIdx.x;
    int b = i >> 12, d = i & (DIM - 1);
    out[i] = Fh[((size_t)b * MEM + idx) * DIM + d];
}

// ---------------------------------------------------------------- host
extern "C" void kernel_launch(void* const* d_in, const int* in_sizes, int n_in,
                              void* d_out, int out_size, void* d_ws, size_t ws_size,
                              hipStream_t stream) {
    const float* x    = (const float*)d_in[0];   // (64, 4096)
    const float* W    = (const float*)d_in[1];   // (4096, 4096)
    const float* bias = (const float*)d_in[2];   // (4096,)
    float* out = (float*)d_out;

    char* base = (char*)d_ws;
    size_t off = 0;
    auto carve = [&](size_t bytes) -> void* {
        void* r = base + off;
        off += (bytes + 255) & ~(size_t)255;
        return r;
    };
    unsigned short* WbT   = (unsigned short*)carve((size_t)DIM * DIM * 2);  // 32 MB bf16 W^T
    float*          Xh    = (float*)carve((size_t)BSZ * MEM * DIM * 4);     // history X
    float*          Fh    = (float*)carve((size_t)BSZ * MEM * DIM * 4);     // history F
    unsigned short* Zb    = (unsigned short*)carve((size_t)BSZ * DIM * 2);  // bf16 GEMM input
    float*          GG    = (float*)carve(BSZ * 25 * 4);
    float*          alpha = (float*)carve(BSZ * MEM * 4);
    float*          pnum  = (float*)carve(512 * 4);
    float*          pden  = (float*)carve(512 * 4);
    int*            flag  = (int*)carve(8);
    int*            lidx  = (int*)carve(8);

    // W -> bf16 transposed (every call; no caching allowed)
    convw_t_kernel<<<dim3(DIM / 64, DIM / 64), 256, 0, stream>>>(W, WbT);
    // F0 = tanh(x + b); seed history, clear flag
    init_kernel<<<(BSZ * DIM) / 256, 256, 0, stream>>>(x, bias, Xh, Fh, Zb, flag, lidx);
    // F1 = f(F0) -> slot 1
    gemm_tanh_kernel<<<DIM / 64, 256, 0, stream>>>(Zb, WbT, x, bias, Fh, 1, flag);

    // Anderson loop, unrolled k = 2..49; converged iterations no-op via device flag
    for (int k = 2; k < 50; ++k) {
        int idx = k % MEM;
        int n = (k < MEM) ? k : MEM;
        gram_kernel<<<BSZ, 256, 0, stream>>>(Xh, Fh, GG, n, flag);
        solve_kernel<<<1, 64, 0, stream>>>(GG, alpha, n, flag);
        update_kernel<<<(BSZ * DIM) / 256, 256, 0, stream>>>(Fh, alpha, Xh, Zb, idx, flag);
        gemm_tanh_kernel<<<DIM / 64, 256, 0, stream>>>(Zb, WbT, x, bias, Fh, idx, flag);
        resid_partial_kernel<<<512, 256, 0, stream>>>(Xh, Fh, pnum, pden, idx, flag);
        resid_final_kernel<<<1, 512, 0, stream>>>(pnum, pden, flag, lidx, idx);
    }

    // output = f(z*) = F[:, last_idx] exactly (Fk = f(Xk) stored in same slot)
    copy_out_kernel<<<(BSZ * DIM) / 256, 256, 0, stream>>>(Fh, lidx, out);
}
